// TemporalMambaBlock_45896020525190
// MI455X (gfx1250) — matrix operand order repaired and checked by reading, not compiled, not run on hardware
//
#include <hip/hip_runtime.h>
#include <hip/hip_bf16.h>
#include <math.h>

typedef _Float16 f16;
typedef __attribute__((ext_vector_type(16))) _Float16 v16h;
typedef __attribute__((ext_vector_type(8)))  _Float16 f16x8;
typedef __attribute__((ext_vector_type(8)))  float    v8f;

#define C_DIM   192
#define D_IN    384
#define D_ST    16
#define DT_RK   12
#define T_LEN   16
#define HW_SZ   784          // 28*28
#define N_SEQ   1568         // 2*28*28

// repacked weight offsets (in f16 elements) inside d_ws
#define W_IN_OFF   0         // 6 kt * 48 nt * 512 = 147456
#define W_X_OFF    147456    // 12 kt * 3 nt * 512 = 18432
#define W_DT_OFF   165888    // 1 kt * 24 nt * 512 = 12288
#define W_OUT_OFF  178176    // 12 kt * 12 nt * 512 = 73728
// total = 251904 f16 = 503808 bytes of d_ws

// ---------------------------------------------------------------------------
// Repack a row-major f32 (K x N) weight matrix into f16 WMMA B-fragments.
// Tile (kt, nt) is 32x16; lane l owns column n = l%16, K = (i/8)*16 + (l/16)*8 + i%8,
// stored as 16 contiguous halfs per lane so the GEMM loop does one 32B load.
// Out-of-range K/N are zero-filled (pads dt_proj K=12->32, x_proj N=44->48).
// ---------------------------------------------------------------------------
__global__ __launch_bounds__(256) void repack_b_kernel(
    const float* __restrict__ src, f16* __restrict__ dst,
    int Kreal, int Nreal, int kTiles, int nTiles) {
  int e = blockIdx.x * 256 + threadIdx.x;
  if (e >= kTiles * nTiles * 512) return;
  int i    = e & 15;
  int lane = (e >> 4) & 31;
  int tile = e >> 9;
  int kt = tile % kTiles;
  int nt = tile / kTiles;
  int nn = nt * 16 + (lane & 15);
  int kk = kt * 32 + (i >> 3) * 16 + (lane >> 4) * 8 + (i & 7);
  float v = (kk < Kreal && nn < Nreal) ? src[(size_t)kk * Nreal + nn] : 0.0f;
  dst[e] = (f16)v;
}

// A fragment (16x32 f16) from row-major LDS: lane l -> row m=l%16,
// halfs K = hf*8..+7 and K = 16+hf*8..+7 (hf = l/16), per ISA 7.12.2.
__device__ __forceinline__ v16h load_a_lds(const f16* base, int stride, int kt, int lane) {
  int m  = lane & 15;
  int hf = lane >> 4;
  const f16* p = base + m * stride + kt * 32 + hf * 8;
  f16x8 lo = *(const f16x8*)(p);
  f16x8 hi = *(const f16x8*)(p + 16);
  v16h a;
#pragma unroll
  for (int i = 0; i < 8; ++i) { a[i] = lo[i]; a[i + 8] = hi[i]; }
  return a;
}

// B fragment (32x16 f16) from repacked global weights: 16 contiguous halfs/lane.
__device__ __forceinline__ v16h load_b_gl(const f16* w, int kTiles, int nt, int kt, int lane) {
  const f16* p = w + (((size_t)nt * kTiles + kt) * 32 + lane) * 16;
  f16x8 lo = *(const f16x8*)(p);
  f16x8 hi = *(const f16x8*)(p + 8);
  v16h b;
#pragma unroll
  for (int i = 0; i < 8; ++i) { b[i] = lo[i]; b[i + 8] = hi[i]; }
  return b;
}

// ---------------------------------------------------------------------------
// Fused Mamba block: one workgroup (8 wave32) per sequence (T=16 rows = WMMA M).
// ---------------------------------------------------------------------------
__global__ __launch_bounds__(256, 2) void mamba_fused_kernel(
    const float* __restrict__ x,      const float* __restrict__ norm_w,
    const float* __restrict__ conv_w, const float* __restrict__ conv_b,
    const float* __restrict__ dt_b,   const float* __restrict__ A_log,
    const float* __restrict__ Dw,
    const f16* __restrict__ w_in,  const f16* __restrict__ w_x,
    const f16* __restrict__ w_dt,  const f16* __restrict__ w_out,
    float* __restrict__ out) {
  __shared__ __align__(16) char smem[65536];
  f16*   s_xn  = (f16*)(smem);            // 16x192 f16  (normed input, A of in_proj)
  f16*   s_u   = (f16*)(smem + 6144);     // 16x384 f16  (u half of xz)
  float* s_dt  = (float*)(smem);          // 16x384 f32  (aliases xn+u, live later)
  f16*   s_z   = (f16*)(smem + 24576);    // 16x384 f16  (z half of xz)
  f16*   s_uc  = (f16*)(smem + 36864);    // 16x384 f16  (post-conv silu, A of x_proj)
  float* s_dbl = (float*)(smem + 49152);  // 16x48  f32  (x_proj out: dt|B|C)
  f16*   s_dtA = (f16*)(smem + 52224);    // 16x32  f16  (dt part padded, A of dt_proj)
  f16*   s_y   = (f16*)(smem + 53248);    // 16x384 f16  (gated scan out, A of out_proj)

  const int tid  = threadIdx.x;
  const int lane = tid & 31;
  const int wave = tid >> 5;
  const int seq  = blockIdx.x;
  const int bb   = seq / HW_SZ;
  const int hw   = seq % HW_SZ;
  // x[b][t][h][w][c] flat = ((b*T + t)*HW + hw)*C + c
  const size_t xbase = ((size_t)bb * T_LEN * HW_SZ + hw) * C_DIM;

  // ---- stage 1: RMSNorm over C per time row -------------------------------
  {
    const int r  = tid >> 4;      // time row 0..15 (16 threads per row)
    const int c0 = tid & 15;
    float vals[12];
    float ss = 0.f;
#pragma unroll
    for (int k = 0; k < 12; ++k) {
      int c = c0 + 16 * k;
      float v = x[xbase + (size_t)r * HW_SZ * C_DIM + c];
      vals[k] = v; ss += v * v;
    }
#pragma unroll
    for (int m = 1; m < 16; m <<= 1) ss += __shfl_xor(ss, m, 32);
    float nrm = rsqrtf(ss * (1.0f / C_DIM) + 1e-6f);
#pragma unroll
    for (int k = 0; k < 12; ++k) {
      int c = c0 + 16 * k;
      s_xn[r * C_DIM + c] = (f16)(vals[k] * nrm * norm_w[c]);
    }
  }
  __syncthreads();

  // ---- stage 2: in_proj (16x192)@(192x768) -> u(16x384), z(16x384) --------
#pragma unroll
  for (int q = 0; q < 6; ++q) {
    int nt = wave * 6 + q;
    v8f acc = {};
#pragma unroll
    for (int kt = 0; kt < 6; ++kt) {
      if (kt + 1 < 6)
        __builtin_prefetch(w_in + (((size_t)nt * 6 + kt + 1) * 32 + lane) * 16, 0, 0);
      v16h av = load_a_lds(s_xn, C_DIM, kt, lane);
      v16h bv = load_b_gl(w_in, 6, nt, kt, lane);
      acc = __builtin_amdgcn_wmma_f32_16x16x32_f16(false, av, false, bv,
                                                   (short)0, acc, false, false);
    }
    f16* dstp = (nt < 24) ? s_u : s_z;
    int nn = ((nt < 24) ? nt : nt - 24) * 16 + (lane & 15);
    int mh = (lane >> 4) * 8;
#pragma unroll
    for (int r = 0; r < 8; ++r)
      dstp[(mh + r) * D_IN + nn] = (f16)acc[r];
  }
  __syncthreads();

  // ---- stage 3: causal depthwise conv (D_CONV=4) + bias + silu ------------
  for (int e = tid; e < T_LEN * D_IN; e += 256) {
    int t = e / D_IN;
    int d = e - t * D_IN;
    float accv = conv_b[d];
#pragma unroll
    for (int k = 0; k < 4; ++k) {
      int tt = t - 3 + k;
      if (tt >= 0) accv += (float)s_u[tt * D_IN + d] * conv_w[d * 4 + k];
    }
    float sg = 1.0f / (1.0f + __expf(-accv));
    s_uc[e] = (f16)(accv * sg);
  }
  __syncthreads();

  // ---- stage 4: x_proj (16x384)@(384x48pad) -> dbl = dt|B|C ----------------
  if (wave < 3) {
    int nt = wave;
    v8f acc = {};
#pragma unroll
    for (int kt = 0; kt < 12; ++kt) {
      v16h av = load_a_lds(s_uc, D_IN, kt, lane);
      v16h bv = load_b_gl(w_x, 12, nt, kt, lane);
      acc = __builtin_amdgcn_wmma_f32_16x16x32_f16(false, av, false, bv,
                                                   (short)0, acc, false, false);
    }
    int nn = nt * 16 + (lane & 15);
    int mh = (lane >> 4) * 8;
#pragma unroll
    for (int r = 0; r < 8; ++r)
      s_dbl[(mh + r) * 48 + nn] = acc[r];
  }
  __syncthreads();

  // pack dt columns (12, zero-padded to 32) as an A fragment source
  for (int e = tid; e < 512; e += 256) {
    int tr = e >> 5, k = e & 31;
    s_dtA[e] = (k < DT_RK) ? (f16)s_dbl[tr * 48 + k] : (f16)0.0f;
  }
  __syncthreads();

  // ---- stage 5: dt_proj (16x32pad)@(32x384) + bias + softplus -------------
#pragma unroll
  for (int q = 0; q < 3; ++q) {
    int nt = wave * 3 + q;
    v16h av = load_a_lds(s_dtA, 32, 0, lane);
    v16h bv = load_b_gl(w_dt, 1, nt, 0, lane);
    v8f acc = {};
    acc = __builtin_amdgcn_wmma_f32_16x16x32_f16(false, av, false, bv,
                                                 (short)0, acc, false, false);
    int nn = nt * 16 + (lane & 15);
    int mh = (lane >> 4) * 8;
    float bias = dt_b[nn];
#pragma unroll
    for (int r = 0; r < 8; ++r) {
      float v = acc[r] + bias;
      float sp = (v > 20.0f) ? v : log1pf(__expf(v));
      s_dt[(mh + r) * D_IN + nn] = sp;   // clobbers s_xn/s_u (dead)
    }
  }
  __syncthreads();

  // ---- stage 6: selective scan, h-state in registers ----------------------
  // thread owns (d, s) pairs: s = tid%16 (state), d = 16*j + tid/16, j=0..23.
  // The 16 lanes sharing a d differ only in s -> 4-step shfl_xor reduces y.
  {
    const int s = tid & 15;
    const int g = tid >> 4;
    float hreg[24], Areg[24], Dreg[24];
#pragma unroll
    for (int j = 0; j < 24; ++j) {
      int d = 16 * j + g;
      hreg[j] = 0.f;
      Areg[j] = -__expf(A_log[d * D_ST + s]);
      Dreg[j] = Dw[d];
    }
    for (int t = 0; t < T_LEN; ++t) {
      float Bt = s_dbl[t * 48 + DT_RK + s];
      float Ct = s_dbl[t * 48 + DT_RK + D_ST + s];
#pragma unroll
      for (int j = 0; j < 24; ++j) {
        int d = 16 * j + g;
        float dtv = s_dt[t * D_IN + d];           // broadcast across the 16 lanes
        float uv  = (float)s_uc[t * D_IN + d];
        hreg[j] = hreg[j] * __expf(dtv * Areg[j]) + dtv * uv * Bt;
        float p = hreg[j] * Ct;
#pragma unroll
        for (int m = 1; m < 16; m <<= 1) p += __shfl_xor(p, m, 32);
        if (s == 0) {
          float yv  = p + uv * Dreg[j];
          float zv  = (float)s_z[t * D_IN + d];
          float sil = zv / (1.0f + __expf(-zv));
          s_y[t * D_IN + d] = (f16)(yv * sil);    // gated output, f16 A operand
        }
      }
    }
  }
  __syncthreads();

  // ---- stage 7: out_proj (16x384)@(384x192) + residual --------------------
  for (int nt = wave; nt < 12; nt += 8) {
    v8f acc = {};
#pragma unroll
    for (int kt = 0; kt < 12; ++kt) {
      if (kt + 1 < 12)
        __builtin_prefetch(w_out + (((size_t)nt * 12 + kt + 1) * 32 + lane) * 16, 0, 0);
      v16h av = load_a_lds(s_y, D_IN, kt, lane);
      v16h bv = load_b_gl(w_out, 12, nt, kt, lane);
      acc = __builtin_amdgcn_wmma_f32_16x16x32_f16(false, av, false, bv,
                                                   (short)0, acc, false, false);
    }
    int c  = nt * 16 + (lane & 15);
    int mh = (lane >> 4) * 8;
#pragma unroll
    for (int r = 0; r < 8; ++r) {
      int t = mh + r;
      size_t gi = xbase + (size_t)t * HW_SZ * C_DIM + c;
      out[gi] = x[gi] + acc[r];
    }
  }
}

extern "C" void kernel_launch(void* const* d_in, const int* in_sizes, int n_in,
                              void* d_out, int out_size, void* d_ws, size_t ws_size,
                              hipStream_t stream) {
  const float* x        = (const float*)d_in[0];
  const float* norm_w   = (const float*)d_in[1];
  const float* in_proj  = (const float*)d_in[2];
  const float* conv_w   = (const float*)d_in[3];
  const float* conv_b   = (const float*)d_in[4];
  const float* x_proj   = (const float*)d_in[5];
  const float* dt_proj  = (const float*)d_in[6];
  const float* dt_pb    = (const float*)d_in[7];
  const float* A_log    = (const float*)d_in[8];
  const float* Dp       = (const float*)d_in[9];
  const float* out_proj = (const float*)d_in[10];
  (void)in_sizes; (void)n_in; (void)out_size; (void)ws_size;

  f16* ws    = (f16*)d_ws;
  f16* w_in  = ws + W_IN_OFF;
  f16* w_x   = ws + W_X_OFF;
  f16* w_dt  = ws + W_DT_OFF;
  f16* w_out = ws + W_OUT_OFF;

  // Repack weights to f16 B-fragment order (L2-resident after first touch).
  repack_b_kernel<<<(147456 + 255) / 256, 256, 0, stream>>>(in_proj,  w_in, 192, 768,  6, 48);
  repack_b_kernel<<<( 18432 + 255) / 256, 256, 0, stream>>>(x_proj,   w_x,  384,  44, 12,  3);
  repack_b_kernel<<<( 12288 + 255) / 256, 256, 0, stream>>>(dt_proj,  w_dt,  12, 384,  1, 24);
  repack_b_kernel<<<( 73728 + 255) / 256, 256, 0, stream>>>(out_proj, w_out, 384, 192, 12, 12);

  mamba_fused_kernel<<<N_SEQ, 256, 0, stream>>>(
      x, norm_w, conv_w, conv_b, dt_pb, A_log, Dp,
      w_in, w_x, w_dt, w_out, (float*)d_out);
}
